// DLRM_Net_5042291605867
// MI455X (gfx1250) — matrix operand order repaired and checked by reading, not compiled, not run on hardware
//
#include <hip/hip_runtime.h>

typedef __attribute__((ext_vector_type(16))) _Float16 v16h;
typedef __attribute__((ext_vector_type(8)))  float    v8f;
typedef __attribute__((ext_vector_type(4)))  unsigned int u32x4;
typedef __attribute__((ext_vector_type(4)))  float    f32x4;
typedef __attribute__((ext_vector_type(4)))  _Float16 h16x4;

union Frag16 { v16h v; u32x4 q[2]; };

#define BATCH   16384
#define NTAB    26
#define EDIM    64
#define NROWS   200000

// ---------------------------------------------------------------------------
// Embedding bag gather+sum (dominant HBM cost): ly[b][t][:] = tab[i0]+tab[i1]
// float4 gathers, f16 writeback (halves the write traffic feeding WMMA).
// ---------------------------------------------------------------------------
__global__ __launch_bounds__(256)
void emb_bag_kernel(const float* __restrict__ tables,
                    const int*   __restrict__ lS_i,
                    _Float16*    __restrict__ ly)
{
    long tid = (long)blockIdx.x * blockDim.x + threadIdx.x;  // BATCH*26*16
    long bag = tid >> 4;            // b*26 + t
    int  c   = (int)(tid & 15);     // which float4 chunk of the 64-dim row
    int  t   = (int)(bag % NTAB);
    long b   = bag / NTAB;
    long i0  = lS_i[(long)t * (BATCH * 2) + 2 * b];
    long i1  = lS_i[(long)t * (BATCH * 2) + 2 * b + 1];
    const f32x4* r0 = (const f32x4*)(tables + ((long)t * NROWS + i0) * EDIM) + c;
    const f32x4* r1 = (const f32x4*)(tables + ((long)t * NROWS + i1) * EDIM) + c;
    f32x4 a = *r0, d = *r1;
    h16x4 o;
    o.x = (_Float16)(a.x + d.x);
    o.y = (_Float16)(a.y + d.y);
    o.z = (_Float16)(a.z + d.z);
    o.w = (_Float16)(a.w + d.w);
    *(h16x4*)(ly + bag * EDIM + c * 4) = o;
}

// ---------------------------------------------------------------------------
// fp32 -> f16 with K zero-padding (weights + dense input). Pads K=13->32 and
// K=415->448 so every GEMM K is a multiple of the WMMA K=32 step.
// ---------------------------------------------------------------------------
__global__ __launch_bounds__(256)
void cvt_pad_f16(const float* __restrict__ src, _Float16* __restrict__ dst,
                 int K, int KP, long total)
{
    long tid = (long)blockIdx.x * blockDim.x + threadIdx.x;
    if (tid >= total) return;
    long n = tid / KP;
    int  k = (int)(tid % KP);
    float v = (k < K) ? src[n * K + k] : 0.0f;
    dst[tid] = (_Float16)v;
}

// ---------------------------------------------------------------------------
// Generic  out = relu(A @ W^T + bias)  tiled WMMA GEMM.
// A: (M,K) f16 row-major,  W: (N,K) f16 row-major, out: (M,N) f16.
// Block tile 128x64, 8 waves; wave w owns rows [16w,16w+16) x 64 cols.
// LDS row stride 40 halves (80B) keeps every fragment read 16B-aligned.
// ---------------------------------------------------------------------------
__global__ __launch_bounds__(256)
void gemm_relu_f16(const _Float16* __restrict__ A, int lda,
                   const _Float16* __restrict__ W, int ldw,
                   const float* __restrict__ bias,
                   _Float16* __restrict__ out, int ldo, int K)
{
    __shared__ _Float16 Al[128 * 40];
    __shared__ _Float16 Wl[64 * 40];
    const int tid  = threadIdx.x;
    const int lane = tid & 31, wv = tid >> 5;
    const int half = lane >> 4, ln = lane & 15;
    const long rowBase = (long)blockIdx.y * 128;
    const int  colBase = blockIdx.x * 64;

    v8f acc[4] = {};

    for (int k0 = 0; k0 < K; k0 += 32) {
        // stage A tile (128x32): 2 x 16B per thread
        #pragma unroll
        for (int j = 0; j < 2; ++j) {
            int idx = tid + j * 256;
            int r = idx >> 2, c = (idx & 3) * 8;
            *(u32x4*)&Al[r * 40 + c] =
                *(const u32x4*)&A[(rowBase + r) * lda + k0 + c];
        }
        // stage W tile (64x32): 1 x 16B per thread
        {
            int r = tid >> 2, c = (tid & 3) * 8;
            *(u32x4*)&Wl[r * 40 + c] =
                *(const u32x4*)&W[((long)colBase + r) * ldw + k0 + c];
        }
        __syncthreads();

        // A fragment (16x32 f16): lane<16 -> K 0..7,16..23 ; lane>=16 -> +8
        Frag16 fa;
        const _Float16* ap = &Al[(wv * 16 + ln) * 40 + 8 * half];
        fa.q[0] = *(const u32x4*)ap;
        fa.q[1] = *(const u32x4*)(ap + 16);

        #pragma unroll
        for (int ns = 0; ns < 4; ++ns) {
            // B fragment (32x16): lane = column, K 0..7,16..23 ; lane>=16 -> +8
            Frag16 fb;
            const _Float16* bp = &Wl[(ns * 16 + ln) * 40 + 8 * half];
            fb.q[0] = *(const u32x4*)bp;
            fb.q[1] = *(const u32x4*)(bp + 16);
            acc[ns] = __builtin_amdgcn_wmma_f32_16x16x32_f16(
                false, fa.v, false, fb.v, (short)0, acc[ns], false, false);
        }
        __syncthreads();
    }

    #pragma unroll
    for (int ns = 0; ns < 4; ++ns) {
        int col = colBase + ns * 16 + ln;
        float bv = bias[col];
        #pragma unroll
        for (int r = 0; r < 8; ++r) {
            long row = rowBase + wv * 16 + r + 8 * half;
            float v = acc[ns][r] + bv;
            v = v > 0.0f ? v : 0.0f;
            out[row * ldo + col] = (_Float16)v;
        }
    }
}

// ---------------------------------------------------------------------------
// Pairwise interaction: per sample T = [x; ly] (27x64, padded to 32x64),
// Z = T.T^T via WMMA (3 of 4 tiles; the (0,1) tile holds only j>i entries),
// R = [x | tril(Z,-1) | zero-pad to 448].  One wave per sample.
// ---------------------------------------------------------------------------
__global__ __launch_bounds__(256)
void interact_kernel(const _Float16* __restrict__ xf,   // B x 64
                     const _Float16* __restrict__ ly,   // B x 26 x 64
                     _Float16* __restrict__ R)          // B x 448
{
    __shared__ _Float16 T[8][32 * 72];   // 72-half row stride, 16B aligned
    const int lane = threadIdx.x & 31, wv = threadIdx.x >> 5;
    const long s = (long)blockIdx.x * 8 + wv;
    _Float16* Ts = T[wv];

    // row 0 = x ; rows 1..26 = embeddings ; rows 27..31 = 0
    *(unsigned int*)&Ts[2 * lane] = *(const unsigned int*)&xf[s * EDIM + 2 * lane];
    const _Float16* lys = ly + s * (NTAB * EDIM);
    #pragma unroll 4
    for (int t = 0; t < NTAB; ++t)
        *(unsigned int*)&Ts[(t + 1) * 72 + 2 * lane] =
            *(const unsigned int*)&lys[t * EDIM + 2 * lane];
    #pragma unroll
    for (int r = 27; r < 32; ++r)
        *(unsigned int*)&Ts[r * 72 + 2 * lane] = 0u;
    __syncthreads();

    const int half = lane >> 4, ln = lane & 15;
    v8f a00 = {}, a10 = {}, a11 = {};
    #pragma unroll
    for (int kk = 0; kk < 64; kk += 32) {
        Frag16 fa0, fa1, fb0, fb1;
        const _Float16* p;
        p = &Ts[(ln) * 72 + kk + 8 * half];
        fa0.q[0] = *(const u32x4*)p; fa0.q[1] = *(const u32x4*)(p + 16);
        p = &Ts[(16 + ln) * 72 + kk + 8 * half];
        fa1.q[0] = *(const u32x4*)p; fa1.q[1] = *(const u32x4*)(p + 16);
        p = &Ts[(ln) * 72 + kk + 8 * half];      // B = T^T: column n = row n of T
        fb0.q[0] = *(const u32x4*)p; fb0.q[1] = *(const u32x4*)(p + 16);
        p = &Ts[(16 + ln) * 72 + kk + 8 * half];
        fb1.q[0] = *(const u32x4*)p; fb1.q[1] = *(const u32x4*)(p + 16);
        a00 = __builtin_amdgcn_wmma_f32_16x16x32_f16(false, fa0.v, false, fb0.v,
                                                     (short)0, a00, false, false);
        a10 = __builtin_amdgcn_wmma_f32_16x16x32_f16(false, fa1.v, false, fb0.v,
                                                     (short)0, a10, false, false);
        a11 = __builtin_amdgcn_wmma_f32_16x16x32_f16(false, fa1.v, false, fb1.v,
                                                     (short)0, a11, false, false);
    }

    _Float16* Rs = R + s * 448;
    *(unsigned int*)&Rs[2 * lane] = *(const unsigned int*)&Ts[2 * lane]; // x copy
    Rs[415 + lane] = (_Float16)0.0f;          // zero-pad cols 415..446
    if (lane == 0) Rs[447] = (_Float16)0.0f;

    auto store_tile = [&](const v8f& acc, int mb, int nb) {
        #pragma unroll
        for (int r = 0; r < 8; ++r) {
            int i = mb + r + 8 * half;
            int j = nb + ln;
            if (i < 27 && j < i)
                Rs[64 + (i * (i - 1)) / 2 + j] = (_Float16)acc[r];
        }
    };
    store_tile(a00, 0, 0);
    store_tile(a10, 16, 0);
    store_tile(a11, 16, 16);
}

// ---------------------------------------------------------------------------
// Final 256 -> 1 layer + sigmoid: wave dot-product + shuffle reduce.
// ---------------------------------------------------------------------------
__global__ __launch_bounds__(256)
void top_final_kernel(const _Float16* __restrict__ act,  // B x 256
                      const _Float16* __restrict__ w2,   // 256
                      const float* __restrict__ b2,      // 1
                      float* __restrict__ out)
{
    const int lane = threadIdx.x & 31, wv = threadIdx.x >> 5;
    const long s = (long)blockIdx.x * 8 + wv;
    union { u32x4 q; _Float16 h[8]; } av, wf;
    av.q = *(const u32x4*)&act[s * 256 + lane * 8];
    wf.q = *(const u32x4*)&w2[lane * 8];
    float sum = 0.0f;
    #pragma unroll
    for (int i = 0; i < 8; ++i) sum += (float)av.h[i] * (float)wf.h[i];
    #pragma unroll
    for (int m = 16; m >= 1; m >>= 1) sum += __shfl_xor(sum, m, 32);
    if (lane == 0) out[s] = 1.0f / (1.0f + __expf(-(sum + b2[0])));
}

// ---------------------------------------------------------------------------
extern "C" void kernel_launch(void* const* d_in, const int* in_sizes, int n_in,
                              void* d_out, int out_size, void* d_ws, size_t ws_size,
                              hipStream_t stream) {
    const float* dense  = (const float*)d_in[0];
    const int*   lS_i   = (const int*)  d_in[1];
    const float* tables = (const float*)d_in[3];
    const float* bW0 = (const float*)d_in[4];  const float* bb0 = (const float*)d_in[5];
    const float* bW1 = (const float*)d_in[6];  const float* bb1 = (const float*)d_in[7];
    const float* bW2 = (const float*)d_in[8];  const float* bb2 = (const float*)d_in[9];
    const float* tW0 = (const float*)d_in[10]; const float* tb0 = (const float*)d_in[11];
    const float* tW1 = (const float*)d_in[12]; const float* tb1 = (const float*)d_in[13];
    const float* tW2 = (const float*)d_in[14]; const float* tb2 = (const float*)d_in[15];
    float* out = (float*)d_out;

    // workspace carve-up (f16 buffers), 256B aligned
    char* base = (char*)d_ws;
    size_t off = 0;
    auto carve = [&](size_t bytes) -> _Float16* {
        _Float16* p = (_Float16*)(base + off);
        off = (off + bytes + 255) & ~(size_t)255;
        return p;
    };
    _Float16* ly    = carve((size_t)BATCH * NTAB * EDIM * 2);  // 54.5 MB
    _Float16* xf    = carve((size_t)BATCH * 64  * 2);
    _Float16* Rbuf  = carve((size_t)BATCH * 448 * 2);
    _Float16* actA  = carve((size_t)BATCH * 512 * 2);
    _Float16* actB  = carve((size_t)BATCH * 256 * 2);
    _Float16* dpad  = carve((size_t)BATCH * 32  * 2);
    _Float16* w0p   = carve((size_t)512 * 32  * 2);
    _Float16* w1h   = carve((size_t)256 * 512 * 2);
    _Float16* w2h   = carve((size_t)64  * 256 * 2);
    _Float16* t0p   = carve((size_t)512 * 448 * 2);
    _Float16* t1h   = carve((size_t)256 * 512 * 2);
    _Float16* t2h   = carve((size_t)256 * 2);

    // 1) embedding bags (HBM-gather bound; the real cost of this net)
    emb_bag_kernel<<<(BATCH * NTAB * 16) / 256, 256, 0, stream>>>(tables, lS_i, ly);

    // 2) f16 conversions / zero-padding
    auto pad = [&](const float* s, _Float16* d, int K, int KP, long N) {
        long tot = N * KP;
        cvt_pad_f16<<<(unsigned)((tot + 255) / 256), 256, 0, stream>>>(s, d, K, KP, tot);
    };
    pad(dense, dpad, 13, 32, BATCH);
    pad(bW0, w0p, 13, 32, 512);
    pad(bW1, w1h, 512, 512, 256);
    pad(bW2, w2h, 256, 256, 64);
    pad(tW0, t0p, 415, 448, 512);
    pad(tW1, t1h, 512, 512, 256);
    pad(tW2, t2h, 256, 256, 1);

    // 3) bottom MLP: 32(pad13)->512->256->64, relu each
    gemm_relu_f16<<<dim3(512 / 64, BATCH / 128), 256, 0, stream>>>(
        dpad, 32, w0p, 32, bb0, actA, 512, 32);
    gemm_relu_f16<<<dim3(256 / 64, BATCH / 128), 256, 0, stream>>>(
        actA, 512, w1h, 512, bb1, actB, 256, 512);
    gemm_relu_f16<<<dim3(64 / 64, BATCH / 128), 256, 0, stream>>>(
        actB, 256, w2h, 256, bb2, xf, 64, 256);

    // 4) pairwise interaction -> R (B x 448, zero-padded past 415)
    interact_kernel<<<BATCH / 8, 256, 0, stream>>>(xf, ly, Rbuf);

    // 5) top MLP: 448(pad415)->512->256, relu each, then 256->1 + sigmoid
    gemm_relu_f16<<<dim3(512 / 64, BATCH / 128), 256, 0, stream>>>(
        Rbuf, 448, t0p, 448, tb0, actA, 512, 448);
    gemm_relu_f16<<<dim3(256 / 64, BATCH / 128), 256, 0, stream>>>(
        actA, 512, t1h, 512, tb1, actB, 256, 512);
    top_final_kernel<<<BATCH / 8, 256, 0, stream>>>(actB, t2h, tb2, out);
}
